// Block_38285338477091
// MI455X (gfx1250) — matrix operand order repaired and checked by reading, not compiled, not run on hardware
//
#include <hip/hip_runtime.h>
#include <math.h>

// ---------------------------------------------------------------------------
// Types
// ---------------------------------------------------------------------------
typedef __attribute__((ext_vector_type(16))) __bf16        v16bf;
typedef __attribute__((ext_vector_type(8)))  float         v8f;
typedef __attribute__((ext_vector_type(4)))  unsigned int  u32x4;

union FragBF {           // 16 bf16 = 32 bytes = two b128 chunks
  u32x4 u[2];
  v16bf v;
};

static __device__ __forceinline__ v8f v8f_zero() {
  v8f z = {0.f, 0.f, 0.f, 0.f, 0.f, 0.f, 0.f, 0.f};
  return z;
}

#define WMMA_BF16(A_, B_, C_) \
  __builtin_amdgcn_wmma_f32_16x16x32_bf16(false, (A_), false, (B_), (short)0, (C_), false, false)

// ---------------------------------------------------------------------------
// Problem constants
// ---------------------------------------------------------------------------
#define Bn   8
#define Tn   1024
#define Cn   1024
#define An   16       // heads
#define Hn   64       // head dim
#define Ln   16       // LoRA rank
#define Mn   4
#define SCALE_LORA (1.0f / 16.0f)
#define EPSLN 1e-5f

// ---------------------------------------------------------------------------
// Weight merge: out_bf16[k][n] = w + (1/L) * sum_r A[k][r] * B[r][n]
// (also used as a plain fp32 -> bf16 converter when la==nullptr)
// ---------------------------------------------------------------------------
__global__ __launch_bounds__(256)
void merge_w_kernel(const float* __restrict__ w,
                    const float* __restrict__ la,   // [K, 16] or null
                    const float* __restrict__ lb,   // [16, N] or null
                    __bf16* __restrict__ out, int K, int N) {
  long idx = (long)blockIdx.x * 256 + threadIdx.x;
  if (idx >= (long)K * N) return;
  int k = (int)(idx / N);
  int n = (int)(idx % N);
  float acc = w[idx];
  if (la != nullptr) {
    float s = 0.f;
#pragma unroll
    for (int r = 0; r < Ln; ++r) s += la[k * Ln + r] * lb[(long)r * N + n];
    acc += s * SCALE_LORA;
  }
  out[idx] = (__bf16)acc;
}

// ---------------------------------------------------------------------------
// LayerNorm: one block per row (C = 1024, 256 threads), bf16 output
// ---------------------------------------------------------------------------
__global__ __launch_bounds__(256)
void ln_kernel(const float* __restrict__ x,
               const float* __restrict__ g,
               const float* __restrict__ b,
               __bf16* __restrict__ out) {
  const int row = blockIdx.x;
  const int tid = threadIdx.x;
  const float* xr = x + (long)row * Cn;
  __shared__ float red[256];

  float xv[4];
#pragma unroll
  for (int k = 0; k < 4; ++k) xv[k] = xr[tid + k * 256];

  float s = xv[0] + xv[1] + xv[2] + xv[3];
  red[tid] = s; __syncthreads();
  for (int off = 128; off > 0; off >>= 1) {
    if (tid < off) red[tid] += red[tid + off];
    __syncthreads();
  }
  const float mu = red[0] * (1.0f / Cn);
  __syncthreads();

  float v = 0.f;
#pragma unroll
  for (int k = 0; k < 4; ++k) { float d = xv[k] - mu; v += d * d; }
  red[tid] = v; __syncthreads();
  for (int off = 128; off > 0; off >>= 1) {
    if (tid < off) red[tid] += red[tid + off];
    __syncthreads();
  }
  const float rstd = rsqrtf(red[0] * (1.0f / Cn) + EPSLN);

#pragma unroll
  for (int k = 0; k < 4; ++k) {
    int i = tid + k * 256;
    out[(long)row * Cn + i] = (__bf16)((xv[k] - mu) * rstd * g[i] + b[i]);
  }
}

// ---------------------------------------------------------------------------
// bf16 WMMA GEMM: out = epilogue( A[M,K] @ W[K,N] )
//   - 256 threads = 8 waves; block tile 128x128, K-step 32
//   - wave (wr, wc) computes rows [wr*32,+32) x cols [wc*64,+64) = 8 WMMA tiles
//   - epilogue: +bias, erf-GELU, +residual, write f32 and/or bf16
// M % 128 == 0, N % 128 == 0, K % 32 == 0 (true for all calls here)
// ---------------------------------------------------------------------------
#define AP 40    // A tile row pitch (bf16): 32 + 8 pad, 80B (16B aligned)
#define BP 136   // B tile row pitch (bf16): 128 + 8 pad, 272B (16B aligned)

__global__ __launch_bounds__(256)
void gemm_bf16_kernel(const __bf16* __restrict__ A,
                      const __bf16* __restrict__ W,
                      const float*  __restrict__ bias,    // [N] or null
                      const float*  __restrict__ resid,   // [M,N] or null
                      float*        __restrict__ outF,    // or null
                      __bf16*       __restrict__ outB,    // or null
                      int M, int N, int K, int doGelu) {
  __shared__ __align__(16) __bf16 As[128 * AP];
  __shared__ __align__(16) __bf16 Bs[32 * BP];

  const int tid  = threadIdx.x;
  const int lane = tid & 31;
  const int wid  = tid >> 5;
  const int wr   = wid & 3;          // wave row group (4 x 32 rows)
  const int wc   = wid >> 2;         // wave col group (2 x 64 cols)
  const int half = lane >> 4;
  const int l16  = lane & 15;
  const long rowBase = (long)blockIdx.y * 128;
  const long colBase = (long)blockIdx.x * 128;

  v8f acc[2][4];
#pragma unroll
  for (int mt = 0; mt < 2; ++mt)
#pragma unroll
    for (int nt = 0; nt < 4; ++nt) acc[mt][nt] = v8f_zero();

  for (int kt = 0; kt < K; kt += 32) {
    // stage A tile: 128 rows x 32 (each row = 4 x b128)
#pragma unroll
    for (int i = 0; i < 2; ++i) {
      int t = tid * 2 + i;                 // 0..511
      int r = t >> 2, seg = t & 3;
      u32x4 dv = *(const u32x4*)(A + (rowBase + r) * K + kt + seg * 8);
      *(u32x4*)(As + r * AP + seg * 8) = dv;
    }
    // stage B tile: 32 rows x 128 (each row = 16 x b128)
#pragma unroll
    for (int i = 0; i < 2; ++i) {
      int t = tid * 2 + i;
      int r = t >> 4, seg = t & 15;
      u32x4 dv = *(const u32x4*)(W + (long)(kt + r) * N + colBase + seg * 8);
      *(u32x4*)(Bs + r * BP + seg * 8) = dv;
    }
    __syncthreads();

    // A fragments: lane = M row (l16), elements = K (half-split per ISA 7.12.2)
    FragBF af[2];
#pragma unroll
    for (int mt = 0; mt < 2; ++mt) {
      int r = wr * 32 + mt * 16 + l16;
      af[mt].u[0] = *(const u32x4*)(As + r * AP + half * 8);
      af[mt].u[1] = *(const u32x4*)(As + r * AP + 16 + half * 8);
    }
    // B fragments: lane = N column (l16), elements = K (half-split, same as A)
    FragBF bf[4];
#pragma unroll
    for (int nt = 0; nt < 4; ++nt) {
      int c = wc * 64 + nt * 16;
#pragma unroll
      for (int s = 0; s < 16; ++s) {
        int kk = (s < 8) ? (half * 8 + s) : (16 + half * 8 + s - 8);
        bf[nt].v[s] = Bs[kk * BP + c + l16];
      }
    }
#pragma unroll
    for (int mt = 0; mt < 2; ++mt)
#pragma unroll
      for (int nt = 0; nt < 4; ++nt)
        acc[mt][nt] = WMMA_BF16(af[mt].v, bf[nt].v, acc[mt][nt]);
    __syncthreads();
  }

  // epilogue (C layout: VGPR r -> row r + half*8; lane l16 -> column)
#pragma unroll
  for (int mt = 0; mt < 2; ++mt) {
#pragma unroll
    for (int nt = 0; nt < 4; ++nt) {
      long gn = colBase + wc * 64 + nt * 16 + l16;
      float bb = (bias != nullptr) ? bias[gn] : 0.f;
#pragma unroll
      for (int r = 0; r < 8; ++r) {
        long gm = rowBase + wr * 32 + mt * 16 + r + half * 8;
        float v = acc[mt][nt][r] + bb;
        if (doGelu) v = 0.5f * v * (1.0f + erff(v * 0.70710678118654752f));
        long gidx = gm * N + gn;
        if (resid != nullptr) v += resid[gidx];
        if (outF != nullptr) outF[gidx] = v;
        if (outB != nullptr) outB[gidx] = (__bf16)v;
      }
    }
  }
}

// ---------------------------------------------------------------------------
// Flash attention (causal), bf16 WMMA.
// qkv: [B*T, 3C] bf16 (q | k | v). out: [B*T, C] bf16.
// grid: (T/128, A, B); block: 256 threads = 8 waves; wave -> 16-query tile.
// Per 32-key chunk: K chunk transposed into LDS (KT[d][key]), V row-major.
// ---------------------------------------------------------------------------
#define KTP 40   // KT pitch (keys) : 32 + 8 pad -> 80B rows
#define VP  72   // V  pitch (dims) : 64 + 8 pad -> 144B rows
#define PP  40   // P  pitch (keys) : 32 + 8 pad

__global__ __launch_bounds__(256)
void attn_kernel(const __bf16* __restrict__ qkv,
                 __bf16* __restrict__ out) {
  const int C3 = 3 * Cn;
  const int b  = blockIdx.z;
  const int h  = blockIdx.y;
  const int Q0 = blockIdx.x * 128;
  const int tid  = threadIdx.x;
  const int lane = tid & 31;
  const int wid  = tid >> 5;
  const int half = lane >> 4;
  const int l16  = lane & 15;
  const int q0w  = Q0 + wid * 16;

  __shared__ __align__(16) __bf16 KT[64 * KTP];
  __shared__ __align__(16) __bf16 Vs[32 * VP];
  __shared__ __align__(16) __bf16 Ps[8 * 16 * PP];

  const long base = (long)b * Tn * C3;

  // Q fragments (A layout): row = q0w + l16, 2 head-dim chunks of 32
  FragBF qa[2];
  {
    const __bf16* qrow = qkv + base + (long)(q0w + l16) * C3 + h * Hn;
#pragma unroll
    for (int c = 0; c < 2; ++c) {
      qa[c].u[0] = *(const u32x4*)(qrow + c * 32 + half * 8);
      qa[c].u[1] = *(const u32x4*)(qrow + c * 32 + 16 + half * 8);
    }
  }

  v8f o[4];
#pragma unroll
  for (int t = 0; t < 4; ++t) o[t] = v8f_zero();
  float mrun[8], lrun[8];
#pragma unroll
  for (int r = 0; r < 8; ++r) { mrun[r] = -1e30f; lrun[r] = 0.f; }

  __bf16* Pw = Ps + wid * 16 * PP;
  const int kcN = (Q0 + 128) / 32;

  for (int kc = 0; kc < kcN; ++kc) {
    // ---- cooperative stage of K (transposed) and V chunks ----
    {
      int key = tid >> 3, seg = tid & 7;            // 32 keys x 8 segments
      long tok = base + (long)(kc * 32 + key) * C3 + h * Hn + seg * 8;
      u32x4 kv = *(const u32x4*)(qkv + tok + Cn);   // K
      const __bf16* kp = (const __bf16*)&kv;
#pragma unroll
      for (int i = 0; i < 8; ++i) KT[(seg * 8 + i) * KTP + key] = kp[i];
      u32x4 vv = *(const u32x4*)(qkv + tok + 2 * Cn);  // V
      *(u32x4*)(Vs + key * VP + seg * 8) = vv;
    }
    __syncthreads();

    if (kc * 32 <= q0w + 15) {   // wave-uniform causal chunk skip
      // ---- scores: S[16 x 32] = Q @ K^T, two 16-key tiles ----
      v8f s[2];
#pragma unroll
      for (int t = 0; t < 2; ++t) {
        FragBF kb0, kb1;    // B layout: lane = key (l16), elems = head-dims (half-split)
#pragma unroll
        for (int s = 0; s < 16; ++s) {
          int kk = (s < 8) ? (half * 8 + s) : (16 + half * 8 + s - 8);
          kb0.v[s] = KT[kk * KTP + t * 16 + l16];
          kb1.v[s] = KT[(32 + kk) * KTP + t * 16 + l16];
        }
        v8f z = v8f_zero();
        s[t] = WMMA_BF16(qa[0].v, kb0.v, z);
        s[t] = WMMA_BF16(qa[1].v, kb1.v, s[t]);
      }
      // ---- scale + causal mask + online softmax ----
      const float sm = 0.125f;   // 1/sqrt(64)
      float p0[8], p1[8], rm[8];
#pragma unroll
      for (int r = 0; r < 8; ++r) {
        int qrow = q0w + r + half * 8;
        int k0 = kc * 32 + l16, k1 = k0 + 16;
        float v0 = (k0 <= qrow) ? s[0][r] * sm : -1e30f;
        float v1 = (k1 <= qrow) ? s[1][r] * sm : -1e30f;
        p0[r] = v0; p1[r] = v1;
        rm[r] = fmaxf(v0, v1);
      }
#pragma unroll
      for (int m = 1; m < 16; m <<= 1)
#pragma unroll
        for (int r = 0; r < 8; ++r) rm[r] = fmaxf(rm[r], __shfl_xor(rm[r], m, 32));

      float alpha[8], rs[8];
#pragma unroll
      for (int r = 0; r < 8; ++r) {
        float mn = fmaxf(mrun[r], rm[r]);
        alpha[r] = expf(mrun[r] - mn);
        mrun[r]  = mn;
        p0[r] = expf(p0[r] - mn);
        p1[r] = expf(p1[r] - mn);
        rs[r] = p0[r] + p1[r];
      }
#pragma unroll
      for (int m = 1; m < 16; m <<= 1)
#pragma unroll
        for (int r = 0; r < 8; ++r) rs[r] += __shfl_xor(rs[r], m, 32);
#pragma unroll
      for (int r = 0; r < 8; ++r) lrun[r] = lrun[r] * alpha[r] + rs[r];
#pragma unroll
      for (int t = 0; t < 4; ++t)
#pragma unroll
        for (int r = 0; r < 8; ++r) o[t][r] *= alpha[r];

      // ---- C layout -> A layout for P via per-wave LDS scratch ----
#pragma unroll
      for (int r = 0; r < 8; ++r) {
        Pw[(r + half * 8) * PP + l16]      = (__bf16)p0[r];
        Pw[(r + half * 8) * PP + 16 + l16] = (__bf16)p1[r];
      }
      asm volatile("s_wait_dscnt 0" ::: "memory");   // wave-local LDS RAW
      FragBF pa;
      pa.u[0] = *(const u32x4*)(Pw + l16 * PP + half * 8);
      pa.u[1] = *(const u32x4*)(Pw + l16 * PP + 16 + half * 8);

      // ---- O += P @ V (4 head-dim tiles of 16) ----
#pragma unroll
      for (int t = 0; t < 4; ++t) {
        FragBF vb;   // lane = head-dim (l16), elems = keys (half-split)
#pragma unroll
        for (int s = 0; s < 16; ++s) {
          int kk = (s < 8) ? (half * 8 + s) : (16 + half * 8 + s - 8);
          vb.v[s] = Vs[kk * VP + t * 16 + l16];
        }
        o[t] = WMMA_BF16(pa.v, vb.v, o[t]);
      }
    }
    __syncthreads();
  }

  // ---- normalize + store [B*T, C] bf16 ----
#pragma unroll
  for (int r = 0; r < 8; ++r) {
    float inv = 1.0f / lrun[r];
    long gm = (long)b * Tn + q0w + r + half * 8;
#pragma unroll
    for (int t = 0; t < 4; ++t)
      out[gm * Cn + h * Hn + t * 16 + l16] = (__bf16)(o[t][r] * inv);
  }
}

// ---------------------------------------------------------------------------
// Launcher
// ---------------------------------------------------------------------------
extern "C" void kernel_launch(void* const* d_in, const int* in_sizes, int n_in,
                              void* d_out, int out_size, void* d_ws, size_t ws_size,
                              hipStream_t stream) {
  (void)in_sizes; (void)n_in; (void)out_size; (void)ws_size;

  const float* btc    = (const float*)d_in[0];
  const float* ln1_g  = (const float*)d_in[1];
  const float* ln1_b  = (const float*)d_in[2];
  const float* qkv_w  = (const float*)d_in[3];
  const float* qkv_A  = (const float*)d_in[4];
  const float* qkv_B  = (const float*)d_in[5];
  const float* proj_w = (const float*)d_in[6];
  const float* ln2_g  = (const float*)d_in[7];
  const float* ln2_b  = (const float*)d_in[8];
  const float* m1_w   = (const float*)d_in[9];
  const float* m1_b   = (const float*)d_in[10];
  const float* m1_A   = (const float*)d_in[11];
  const float* m1_B   = (const float*)d_in[12];
  const float* m2_w   = (const float*)d_in[13];
  const float* m2_b   = (const float*)d_in[14];
  const float* m2_A   = (const float*)d_in[15];
  const float* m2_B   = (const float*)d_in[16];

  char* ws = (char*)d_ws;
  size_t cur = 0;
  auto alloc = [&](size_t bytes) {
    size_t o = cur;
    cur += (bytes + 255) & ~(size_t)255;
    return o;
  };
  const long NT = (long)Bn * Tn;  // 8192 tokens
  __bf16* qkv_eff  = (__bf16*)(ws + alloc((size_t)Cn * 3 * Cn * 2));       //  6 MB
  __bf16* proj_eff = (__bf16*)(ws + alloc((size_t)Cn * Cn * 2));           //  2 MB
  __bf16* m1_eff   = (__bf16*)(ws + alloc((size_t)Cn * Mn * Cn * 2));      //  8 MB
  __bf16* m2_eff   = (__bf16*)(ws + alloc((size_t)Mn * Cn * Cn * 2));      //  8 MB
  __bf16* xln      = (__bf16*)(ws + alloc((size_t)NT * Cn * 2));           // 16 MB (reused as attn_out)
  __bf16* big      = (__bf16*)(ws + alloc((size_t)NT * Mn * Cn * 2));      // 64 MB (qkv acts, then h)
  float*  res1     = (float*) (ws + alloc((size_t)NT * Cn * 4));           // 32 MB
  __bf16* yln      = (__bf16*)(ws + alloc((size_t)NT * Cn * 2));           // 16 MB

  // 1) fold LoRA into weights, convert to bf16
  merge_w_kernel<<<(Cn * 3 * Cn + 255) / 256, 256, 0, stream>>>(qkv_w, qkv_A, qkv_B, qkv_eff, Cn, 3 * Cn);
  merge_w_kernel<<<(Cn * Cn + 255) / 256, 256, 0, stream>>>(proj_w, nullptr, nullptr, proj_eff, Cn, Cn);
  merge_w_kernel<<<(Cn * Mn * Cn + 255) / 256, 256, 0, stream>>>(m1_w, m1_A, m1_B, m1_eff, Cn, Mn * Cn);
  merge_w_kernel<<<(Mn * Cn * Cn + 255) / 256, 256, 0, stream>>>(m2_w, m2_A, m2_B, m2_eff, Mn * Cn, Cn);

  // 2) LN1 -> bf16
  ln_kernel<<<(int)NT, 256, 0, stream>>>(btc, ln1_g, ln1_b, xln);

  // 3) QKV GEMM: [8192,1024] x [1024,3072] -> bf16 acts
  gemm_bf16_kernel<<<dim3(3 * Cn / 128, (int)NT / 128), 256, 0, stream>>>(
      xln, qkv_eff, nullptr, nullptr, nullptr, big, (int)NT, 3 * Cn, Cn, 0);

  // 4) causal flash attention -> bf16 (reuse xln buffer)
  attn_kernel<<<dim3(Tn / 128, An, Bn), 256, 0, stream>>>(big, xln);

  // 5) proj GEMM + residual(btc) -> res1 (fp32)
  gemm_bf16_kernel<<<dim3(Cn / 128, (int)NT / 128), 256, 0, stream>>>(
      xln, proj_eff, nullptr, btc, res1, nullptr, (int)NT, Cn, Cn, 0);

  // 6) LN2 -> bf16
  ln_kernel<<<(int)NT, 256, 0, stream>>>(res1, ln2_g, ln2_b, yln);

  // 7) MLP1 GEMM + bias + erf-GELU -> h (bf16, reuse big)
  gemm_bf16_kernel<<<dim3(Mn * Cn / 128, (int)NT / 128), 256, 0, stream>>>(
      yln, m1_eff, m1_b, nullptr, nullptr, big, (int)NT, Mn * Cn, Cn, 1);

  // 8) MLP2 GEMM + bias + residual(res1) -> final fp32 output
  gemm_bf16_kernel<<<dim3(Cn / 128, (int)NT / 128), 256, 0, stream>>>(
      big, m2_eff, m2_b, res1, (float*)d_out, nullptr, (int)NT, Cn, Mn * Cn, 0);
}
